// Model_79972291051986
// MI455X (gfx1250) — matrix operand order repaired and checked by reading, not compiled, not run on hardware
//
#include <hip/hip_runtime.h>
#include <hip/hip_bf16.h>
#include <math.h>

// ---------------------------------------------------------------------------
// Types for CDNA5 WMMA (wave32): v_wmma_f32_16x16x32_bf16
// ---------------------------------------------------------------------------
typedef __bf16 bf16_t;
typedef __attribute__((ext_vector_type(16))) __bf16        v16bf;
typedef __attribute__((ext_vector_type(8)))  float         v8f;
typedef __attribute__((ext_vector_type(4)))  unsigned int  u32x4;

union BF16x16 {
    u32x4  u4[2];
    v16bf  v;
    bf16_t h[16];
};

static __device__ __forceinline__ v8f wmma_bf16(v16bf a, v16bf b, v8f c) {
    // 8 args: (neg_a, A, neg_b, B, c_mod, C, reuse_a, reuse_b)
    return __builtin_amdgcn_wmma_f32_16x16x32_bf16(false, a, false, b,
                                                   (short)0, c, false, false);
}

// ---------------------------------------------------------------------------
// Problem constants
// ---------------------------------------------------------------------------
#define DEPTH   3
#define HEADS   8
#define DIM     128
#define MLP_D   4096
#define N_TOK   125
#define BATCH   8
#define NTOKS   625            // N
#define NPAD    640            // N padded to 16
#define MREAL   5000           // B*N rows
#define MPAD    5008           // padded to 16
#define MTILES  313            // MPAD/16
#define SCALE_Q 0.25f          // (DIM/HEADS)^-0.5 = 16^-0.5

// ---------------------------------------------------------------------------
// WMMA GEMM mainloop: one wave computes a 16(M) x 64(N) strip of A @ Wt^T,
// 4 accumulators sharing one A tile per K chunk (4x A reuse).
// Software double-buffered: next chunk's A + 4xB loads are issued before the
// current chunk's 4 WMMAs execute, so loads overlap the matrix pipe.
//   A : (MPAD x K) bf16 row-major
//   Wt: (Nc  x K) bf16 row-major (weights pre-transposed so B loads are
//       contiguous in K per lane)
// A layout per ISA 7.12.2 (16-bit A 16x32): lane L -> row M = L&15,
//   K runs of 8 at {0,16} (lanes 0-15) / {8,24} (lanes 16-31).
// B layout (32x16, KxN): lane L -> col N = L&15, K runs of 8 as for A:
//   lanes 0-15 hold K=0..7,16..23, lanes 16-31 hold K=8..15,24..31.
// ---------------------------------------------------------------------------
template <int K>
static __device__ __forceinline__ void gemm_strip4(const bf16_t* __restrict__ A,
                                                   const bf16_t* __restrict__ Wt,
                                                   int m, int n0, int lane,
                                                   v8f acc[4]) {
    const int aoff = (lane < 16) ? 0 : 8;
    const int boff = (lane >> 4) << 3;
    const bf16_t* arow  = A  + (size_t)m * K;
    const bf16_t* brow0 = Wt + (size_t)(n0 + (lane & 15)) * K;

    BF16x16 a_cur, b_cur[4];
    a_cur.u4[0] = *(const u32x4*)(arow + aoff);
    a_cur.u4[1] = *(const u32x4*)(arow + aoff + 16);
#pragma unroll
    for (int j = 0; j < 4; j++) {
        const bf16_t* br = brow0 + (size_t)j * 16 * K;
        b_cur[j].u4[0] = *(const u32x4*)(br + boff);
        b_cur[j].u4[1] = *(const u32x4*)(br + boff + 16);
    }

#pragma unroll 4
    for (int kk = 32; kk < K; kk += 32) {
        BF16x16 a_nxt, b_nxt[4];
        a_nxt.u4[0] = *(const u32x4*)(arow + kk + aoff);
        a_nxt.u4[1] = *(const u32x4*)(arow + kk + aoff + 16);
#pragma unroll
        for (int j = 0; j < 4; j++) {
            const bf16_t* br = brow0 + (size_t)j * 16 * K;
            b_nxt[j].u4[0] = *(const u32x4*)(br + kk + boff);
            b_nxt[j].u4[1] = *(const u32x4*)(br + kk + boff + 16);
        }
#pragma unroll
        for (int j = 0; j < 4; j++)
            acc[j] = wmma_bf16(a_cur.v, b_cur[j].v, acc[j]);
        a_cur = a_nxt;
#pragma unroll
        for (int j = 0; j < 4; j++) b_cur[j] = b_nxt[j];
    }
#pragma unroll
    for (int j = 0; j < 4; j++)
        acc[j] = wmma_bf16(a_cur.v, b_cur[j].v, acc[j]);
}

// ---------------------------------------------------------------------------
// Small utility kernels
// ---------------------------------------------------------------------------
__global__ void copy_f32_kernel(const float* __restrict__ s,
                                float* __restrict__ d, int n) {
    int i = blockIdx.x * blockDim.x + threadIdx.x;
    if (i < n) d[i] = s[i];
}

// fp32 (K x N) -> bf16 transposed (N x K)
__global__ void wtrans_kernel(const float* __restrict__ src,
                              bf16_t* __restrict__ dst, int K, int N) {
    int idx = blockIdx.x * blockDim.x + threadIdx.x;
    if (idx >= K * N) return;
    int k = idx / N, n = idx % N;
    dst[(size_t)n * K + k] = (bf16_t)src[idx];
}

// Column mask bias: 0 if attendable, -1e30 otherwise (incl. padded cols)
__global__ void colbias_kernel(const int* __restrict__ mask,
                               float* __restrict__ cb) {
    int idx = blockIdx.x * blockDim.x + threadIdx.x;
    if (idx >= BATCH * NPAD) return;
    int b = idx / NPAD, c = idx % NPAD;
    bool allowed;
    if (c < 4 * N_TOK)      allowed = mask[b * 4 + c / N_TOK] != 0;
    else                    allowed = (c < NTOKS);     // fusion block / padding
    cb[idx] = allowed ? 0.f : -1e30f;
}

// LayerNorm over DIM=128, emits bf16 rows for the GEMM A operand
__global__ void ln_kernel(const float* __restrict__ x,
                          const float* __restrict__ g,
                          const float* __restrict__ bta,
                          bf16_t* __restrict__ h) {
    __shared__ float red[DIM];
    int row = blockIdx.x;          // 0..MREAL-1
    int tid = threadIdx.x;         // 0..127
    float v = x[(size_t)row * DIM + tid];
    red[tid] = v;
    __syncthreads();
#pragma unroll
    for (int s = DIM / 2; s > 0; s >>= 1) {
        if (tid < s) red[tid] += red[tid + s];
        __syncthreads();
    }
    float mu = red[0] * (1.f / DIM);
    __syncthreads();
    float d = v - mu;
    red[tid] = d * d;
    __syncthreads();
#pragma unroll
    for (int s = DIM / 2; s > 0; s >>= 1) {
        if (tid < s) red[tid] += red[tid + s];
        __syncthreads();
    }
    float var = red[0] * (1.f / DIM);
    float y = d * rsqrtf(var + 1e-5f) * g[tid] + bta[tid];
    h[(size_t)row * DIM + tid] = (bf16_t)y;
}

// ---------------------------------------------------------------------------
// GEMM: qkv = h @ Wqkv  (K=128, Nc=384); scatter into q/k/vT bf16 buffers.
//   q: (B,H,NPAD,32)  head-dim zero-padded to 32, scaled by SCALE_Q
//   k: (B,H,NPAD,32)  head-dim zero-padded to 32
//   vT:(B,H,16,NPAD)  transposed so P@V B-matrix loads are contiguous
// ---------------------------------------------------------------------------
__global__ void gemm_qkv_kernel(const bf16_t* __restrict__ A,
                                const bf16_t* __restrict__ Wt,
                                bf16_t* __restrict__ qb,
                                bf16_t* __restrict__ kb,
                                bf16_t* __restrict__ vtb) {
    const int Nstrips = 384 / 64;                 // 6
    int wid = blockIdx.x * (blockDim.x >> 5) + (threadIdx.x >> 5);
    if (wid >= MTILES * Nstrips) return;
    int mt = wid / Nstrips, ns = wid % Nstrips;
    int lane = threadIdx.x & 31;
    int lo = lane & 15, hi = lane >> 4;
    v8f acc[4] = {{0.f, 0.f, 0.f, 0.f, 0.f, 0.f, 0.f, 0.f},
                  {0.f, 0.f, 0.f, 0.f, 0.f, 0.f, 0.f, 0.f},
                  {0.f, 0.f, 0.f, 0.f, 0.f, 0.f, 0.f, 0.f},
                  {0.f, 0.f, 0.f, 0.f, 0.f, 0.f, 0.f, 0.f}};
    gemm_strip4<DIM>(A, Wt, mt * 16 + lo, ns * 64, lane, acc);
#pragma unroll
    for (int r = 0; r < 8; r++) {
        int row = mt * 16 + hi * 8 + r;
        if (row < MREAL) {
            int b = row / NTOKS, t = row % NTOKS;
#pragma unroll
            for (int j = 0; j < 4; j++) {
                int col = ns * 64 + j * 16 + lo;
                float v = acc[j][r];
                if (col < 128) {
                    int hh = col >> 4, d = col & 15;
                    qb[(((size_t)(b * HEADS + hh)) * NPAD + t) * 32 + d] =
                        (bf16_t)(v * SCALE_Q);
                } else if (col < 256) {
                    int c2 = col - 128, hh = c2 >> 4, d = c2 & 15;
                    kb[(((size_t)(b * HEADS + hh)) * NPAD + t) * 32 + d] =
                        (bf16_t)v;
                } else {
                    int c2 = col - 256, hh = c2 >> 4, d = c2 & 15;
                    vtb[(((size_t)(b * HEADS + hh)) * 16 + d) * NPAD + t] =
                        (bf16_t)v;
                }
            }
        }
    }
}

// ---------------------------------------------------------------------------
// GEMM with residual epilogue: xbuf[m][c] += acc + bias[c]   (Nc = 128)
// Used for attention proj (K=128) and MLP down-proj (K=4096).
// ---------------------------------------------------------------------------
template <int K>
__global__ void gemm_xres_kernel(const bf16_t* __restrict__ A,
                                 const bf16_t* __restrict__ Wt,
                                 const float* __restrict__ bias,
                                 float* __restrict__ xbuf) {
    const int Nstrips = DIM / 64;                 // 2
    int wid = blockIdx.x * (blockDim.x >> 5) + (threadIdx.x >> 5);
    if (wid >= MTILES * Nstrips) return;
    int mt = wid / Nstrips, ns = wid % Nstrips;
    int lane = threadIdx.x & 31;
    int lo = lane & 15, hi = lane >> 4;
    v8f acc[4] = {{0.f, 0.f, 0.f, 0.f, 0.f, 0.f, 0.f, 0.f},
                  {0.f, 0.f, 0.f, 0.f, 0.f, 0.f, 0.f, 0.f},
                  {0.f, 0.f, 0.f, 0.f, 0.f, 0.f, 0.f, 0.f},
                  {0.f, 0.f, 0.f, 0.f, 0.f, 0.f, 0.f, 0.f}};
    gemm_strip4<K>(A, Wt, mt * 16 + lo, ns * 64, lane, acc);
    float bc[4];
#pragma unroll
    for (int j = 0; j < 4; j++) bc[j] = bias[ns * 64 + j * 16 + lo];
#pragma unroll
    for (int r = 0; r < 8; r++) {
        int row = mt * 16 + hi * 8 + r;
        if (row < MREAL) {
#pragma unroll
            for (int j = 0; j < 4; j++) {
                size_t o = (size_t)row * DIM + ns * 64 + j * 16 + lo;
                xbuf[o] = xbuf[o] + acc[j][r] + bc[j];
            }
        }
    }
}

// ---------------------------------------------------------------------------
// GEMM MLP up-proj: gelu(h2 @ W1f + b1f) -> bf16   (K=128, Nc=4096)
// ---------------------------------------------------------------------------
__global__ void gemm_mlp1_kernel(const bf16_t* __restrict__ A,
                                 const bf16_t* __restrict__ Wt,
                                 const float* __restrict__ bias,
                                 bf16_t* __restrict__ mlp) {
    const int Nstrips = MLP_D / 64;               // 64
    int wid = blockIdx.x * (blockDim.x >> 5) + (threadIdx.x >> 5);
    if (wid >= MTILES * Nstrips) return;
    int mt = wid / Nstrips, ns = wid % Nstrips;
    int lane = threadIdx.x & 31;
    int lo = lane & 15, hi = lane >> 4;
    v8f acc[4] = {{0.f, 0.f, 0.f, 0.f, 0.f, 0.f, 0.f, 0.f},
                  {0.f, 0.f, 0.f, 0.f, 0.f, 0.f, 0.f, 0.f},
                  {0.f, 0.f, 0.f, 0.f, 0.f, 0.f, 0.f, 0.f},
                  {0.f, 0.f, 0.f, 0.f, 0.f, 0.f, 0.f, 0.f}};
    gemm_strip4<DIM>(A, Wt, mt * 16 + lo, ns * 64, lane, acc);
    float bc[4];
#pragma unroll
    for (int j = 0; j < 4; j++) bc[j] = bias[ns * 64 + j * 16 + lo];
#pragma unroll
    for (int r = 0; r < 8; r++) {
        int row = mt * 16 + hi * 8 + r;
        if (row < MREAL) {
#pragma unroll
            for (int j = 0; j < 4; j++) {
                float u = acc[j][r] + bc[j];
                float ge = 0.5f * u * (1.f + erff(u * 0.70710678118654752f));
                mlp[(size_t)row * MLP_D + ns * 64 + j * 16 + lo] = (bf16_t)ge;
            }
        }
    }
}

// ---------------------------------------------------------------------------
// Attention: one wave per (b, h, 16-row tile). Scores strip in LDS,
// masked softmax, P@V via WMMA, write o (bf16, head-concatenated).
// ---------------------------------------------------------------------------
__global__ void attn_kernel(const bf16_t* __restrict__ qb,
                            const bf16_t* __restrict__ kb,
                            const bf16_t* __restrict__ vtb,
                            const float* __restrict__ colbias,
                            bf16_t* __restrict__ ob) {
    __shared__ float sc[16 * NPAD];   // 40 KB score strip
    __shared__ float red[64];
    __shared__ float inv[16];

    int rt = blockIdx.x, h = blockIdx.y, b = blockIdx.z;
    int lane = threadIdx.x & 31;
    int lo = lane & 15, hi = lane >> 4;

    const bf16_t* qbase = qb  + ((size_t)(b * HEADS + h)) * NPAD * 32;
    const bf16_t* kbase = kb  + ((size_t)(b * HEADS + h)) * NPAD * 32;
    const bf16_t* vbase = vtb + ((size_t)(b * HEADS + h)) * 16 * NPAD;
    const float*  cb    = colbias + b * NPAD;

    // q tile (A operand), loaded once; scale already folded in
    int aoff = hi ? 8 : 0;
    int qtok = rt * 16 + lo;
    BF16x16 aq;
    aq.u4[0] = *(const u32x4*)(qbase + (size_t)qtok * 32 + aoff);
    aq.u4[1] = *(const u32x4*)(qbase + (size_t)qtok * 32 + aoff + 16);

    // --- phase 1: scores strip (16 x NPAD) into LDS, + mask bias -----------
#pragma unroll 2
    for (int ct = 0; ct < NPAD / 16; ct++) {
        int ktok = ct * 16 + lo;
        BF16x16 bk;
        int boff = hi << 3;
        bk.u4[0] = *(const u32x4*)(kbase + (size_t)ktok * 32 + boff);
        bk.u4[1] = *(const u32x4*)(kbase + (size_t)ktok * 32 + boff + 16);
        v8f s = {0.f, 0.f, 0.f, 0.f, 0.f, 0.f, 0.f, 0.f};
        s = wmma_bf16(aq.v, bk.v, s);
        float bias = cb[ct * 16 + lo];
#pragma unroll
        for (int r = 0; r < 8; r++)
            sc[(hi * 8 + r) * NPAD + ct * 16 + lo] = s[r] + bias;
    }
    __syncthreads();

    // --- phase 2: softmax (2 lanes per row, 320 cols each) -----------------
    {
        int base = lo * NPAD + hi * 320;
        float mx = -3.0e38f;
        for (int i = 0; i < 320; i++) mx = fmaxf(mx, sc[base + i]);
        red[hi * 16 + lo] = mx;
        __syncthreads();
        float M = fmaxf(red[lo], red[16 + lo]);
        float sum = 0.f;
        for (int i = 0; i < 320; i++) {
            float e = __expf(sc[base + i] - M);
            sc[base + i] = e;
            sum += e;
        }
        red[32 + hi * 16 + lo] = sum;
        __syncthreads();
        if (lane < 16) inv[lane] = 1.f / (red[32 + lane] + red[48 + lane]);
        __syncthreads();
    }

    // --- phase 3: O = P @ V via WMMA (20 chunks of K=32 tokens) ------------
    v8f acc = {0.f, 0.f, 0.f, 0.f, 0.f, 0.f, 0.f, 0.f};
#pragma unroll 2
    for (int cc = 0; cc < NPAD / 32; cc++) {
        BF16x16 ap, bv;
        int kb0 = cc * 32 + aoff;
#pragma unroll
        for (int e = 0; e < 8; e++) ap.h[e]     = (bf16_t)sc[lo * NPAD + kb0 + e];
#pragma unroll
        for (int e = 0; e < 8; e++) ap.h[8 + e] = (bf16_t)sc[lo * NPAD + kb0 + 16 + e];
        int kb1 = cc * 32 + (hi << 3);
        bv.u4[0] = *(const u32x4*)(vbase + (size_t)lo * NPAD + kb1);
        bv.u4[1] = *(const u32x4*)(vbase + (size_t)lo * NPAD + kb1 + 16);
        acc = wmma_bf16(ap.v, bv.v, acc);
    }

#pragma unroll
    for (int r = 0; r < 8; r++) {
        int rr = hi * 8 + r;
        int tok = rt * 16 + rr;
        if (tok < NTOKS) {
            float val = acc[r] * inv[rr];
            ob[((size_t)(b * NTOKS + tok)) * DIM + h * 16 + lo] = (bf16_t)val;
        }
    }
}

// ---------------------------------------------------------------------------
// Host-side orchestration
// ---------------------------------------------------------------------------
static inline size_t align256(size_t x) { return (x + 255) & ~(size_t)255; }

extern "C" void kernel_launch(void* const* d_in, const int* in_sizes, int n_in,
                              void* d_out, int out_size, void* d_ws, size_t ws_size,
                              hipStream_t stream) {
    const float* x_in  = (const float*)d_in[0];
    const int*   mask  = (const int*)  d_in[1];
    const float* Wqkv  = (const float*)d_in[2];
    const float* Wproj = (const float*)d_in[3];
    const float* bproj = (const float*)d_in[4];
    const float* g1    = (const float*)d_in[5];
    const float* b1    = (const float*)d_in[6];
    const float* g2    = (const float*)d_in[7];
    const float* b2    = (const float*)d_in[8];
    const float* W1f   = (const float*)d_in[9];
    const float* b1f   = (const float*)d_in[10];
    const float* W2f   = (const float*)d_in[11];
    const float* b2f   = (const float*)d_in[12];

    // workspace layout (bytes)
    char* ws = (char*)d_ws;
    size_t off = 0;
    float*  xbuf   = (float*) (ws + off); off = align256(off + (size_t)MPAD * DIM * 4);
    bf16_t* hbuf   = (bf16_t*)(ws + off); off = align256(off + (size_t)MPAD * DIM * 2);
    bf16_t* wqkvT  = (bf16_t*)(ws + off); off = align256(off + (size_t)DEPTH * 384 * DIM * 2);
    bf16_t* wprojT = (bf16_t*)(ws + off); off = align256(off + (size_t)DEPTH * DIM * DIM * 2);
    bf16_t* w1T    = (bf16_t*)(ws + off); off = align256(off + (size_t)DEPTH * MLP_D * DIM * 2);
    bf16_t* w2T    = (bf16_t*)(ws + off); off = align256(off + (size_t)DEPTH * DIM * MLP_D * 2);
    bf16_t* qbuf   = (bf16_t*)(ws + off); off = align256(off + (size_t)BATCH * HEADS * NPAD * 32 * 2);
    bf16_t* kbuf   = (bf16_t*)(ws + off); off = align256(off + (size_t)BATCH * HEADS * NPAD * 32 * 2);
    bf16_t* vtbuf  = (bf16_t*)(ws + off); off = align256(off + (size_t)BATCH * HEADS * 16 * NPAD * 2);
    bf16_t* obuf   = (bf16_t*)(ws + off); off = align256(off + (size_t)MPAD * DIM * 2);
    float*  cbuf   = (float*) (ws + off); off = align256(off + (size_t)BATCH * NPAD * 4);
    bf16_t* mlpbuf = (bf16_t*)(ws + off); off = align256(off + (size_t)MPAD * MLP_D * 2);
    size_t total_ws = off;

    // zero-fill once per launch (padding regions must be zero)
    hipMemsetAsync(d_ws, 0, total_ws, stream);

    // convert + transpose all weights to bf16
    for (int l = 0; l < DEPTH; l++) {
        wtrans_kernel<<<(DIM * 384 + 255) / 256, 256, 0, stream>>>(
            Wqkv + (size_t)l * DIM * 384, wqkvT + (size_t)l * 384 * DIM, DIM, 384);
        wtrans_kernel<<<(DIM * DIM + 255) / 256, 256, 0, stream>>>(
            Wproj + (size_t)l * DIM * DIM, wprojT + (size_t)l * DIM * DIM, DIM, DIM);
        wtrans_kernel<<<(DIM * MLP_D + 255) / 256, 256, 0, stream>>>(
            W1f + (size_t)l * DIM * MLP_D, w1T + (size_t)l * MLP_D * DIM, DIM, MLP_D);
        wtrans_kernel<<<(MLP_D * DIM + 255) / 256, 256, 0, stream>>>(
            W2f + (size_t)l * MLP_D * DIM, w2T + (size_t)l * DIM * MLP_D, MLP_D, DIM);
    }

    // residual stream init + column mask bias
    copy_f32_kernel<<<(MREAL * DIM + 255) / 256, 256, 0, stream>>>(x_in, xbuf, MREAL * DIM);
    colbias_kernel<<<(BATCH * NPAD + 255) / 256, 256, 0, stream>>>(mask, cbuf);

    const int WPB = 8; // waves per 256-thread block
    for (int l = 0; l < DEPTH; l++) {
        // attention block
        ln_kernel<<<MREAL, DIM, 0, stream>>>(xbuf, g1 + l * DIM, b1 + l * DIM, hbuf);
        {
            int strips = MTILES * (384 / 64);
            gemm_qkv_kernel<<<(strips + WPB - 1) / WPB, 256, 0, stream>>>(
                hbuf, wqkvT + (size_t)l * 384 * DIM, qbuf, kbuf, vtbuf);
        }
        attn_kernel<<<dim3(NPAD / 16, HEADS, BATCH), 32, 0, stream>>>(
            qbuf, kbuf, vtbuf, cbuf, obuf);
        {
            int strips = MTILES * (DIM / 64);
            gemm_xres_kernel<DIM><<<(strips + WPB - 1) / WPB, 256, 0, stream>>>(
                obuf, wprojT + (size_t)l * DIM * DIM, bproj + l * DIM, xbuf);
        }
        // MLP block
        ln_kernel<<<MREAL, DIM, 0, stream>>>(xbuf, g2 + l * DIM, b2 + l * DIM, hbuf);
        {
            int strips = MTILES * (MLP_D / 64);
            gemm_mlp1_kernel<<<(strips + WPB - 1) / WPB, 256, 0, stream>>>(
                hbuf, w1T + (size_t)l * MLP_D * DIM, b1f + l * MLP_D, mlpbuf);
        }
        {
            int strips = MTILES * (DIM / 64);
            gemm_xres_kernel<MLP_D><<<(strips + WPB - 1) / WPB, 256, 0, stream>>>(
                mlpbuf, w2T + (size_t)l * DIM * MLP_D, b2f + l * DIM, xbuf);
        }
    }

    copy_f32_kernel<<<(MREAL * DIM + 255) / 256, 256, 0, stream>>>(
        xbuf, (float*)d_out, MREAL * DIM);
}
